// Decoder_31860067401767
// MI455X (gfx1250) — matrix operand order repaired and checked by reading, not compiled, not run on hardware
//
#include <hip/hip_runtime.h>
#include <math.h>

#define V_ 32000
#define H_ 1024
#define S_ 512
#define B_ 16
#define T_ 128

typedef unsigned short u16;
typedef __attribute__((ext_vector_type(16))) __bf16 v16bf;
typedef __attribute__((ext_vector_type(8)))  float  v8f;
typedef __attribute__((ext_vector_type(4)))  unsigned int u32x4;

union V16 { v16bf v; u32x4 q[2]; };

__device__ inline u16 f2bf(float f) {
  unsigned u = __float_as_uint(f);
  u += 0x7FFFu + ((u >> 16) & 1u);           // round-to-nearest-even
  return (u16)(u >> 16);
}
__device__ inline float bf2f(u16 x) { return __uint_as_float(((unsigned)x) << 16); }
__device__ inline float sigm(float x) { return 1.f / (1.f + __expf(-x)); }

// ---------------------------------------------------------------------------
// NT adjacent 16x16 output tiles sharing one A fragment, with EXPLICIT
// double-buffered K pipeline (2x-unrolled, rotating fragment registers):
//   stage i issues loads for K-chunk i+1, then WMMAs on K-chunk i.
// Every v_wmma therefore depends only on loads issued one stage back while
// 2+2*NT newer loads remain outstanding -> compiler must emit partial
// s_wait_loadcnt instead of wait-to-zero.
// A row-major [M,K] bf16, W row-major [N,K] bf16 (B = W^T); CDNA5 16-bit
// fragment layouts map to contiguous 16B loads for both operands.
// K must be a multiple of 64. The final prefetch reads one 64B fragment past
// row K; all WMMA operand buffers are followed by more workspace, so safe.
// ---------------------------------------------------------------------------
template <int NT>
__device__ inline void wmma_mt(const u16* __restrict__ A, int lda,
                               const u16* __restrict__ W, int ldw,
                               int m0, int n0, int K, v8f acc[NT]) {
  const int lane = threadIdx.x & 31;
  const int hi   = lane >> 4;
  const int l    = lane & 15;
  const u16* ar = A + (size_t)(m0 + l) * lda + hi * 8;
  const u16* wr[NT];
  #pragma unroll
  for (int t = 0; t < NT; ++t)
    wr[t] = W + (size_t)(n0 + t * 16 + l) * ldw + hi * 8;

  V16 a0, a1, b0[NT], b1[NT];
  a0.q[0] = *(const u32x4*)(ar);
  a0.q[1] = *(const u32x4*)(ar + 16);
  #pragma unroll
  for (int t = 0; t < NT; ++t) {
    b0[t].q[0] = *(const u32x4*)(wr[t]);
    b0[t].q[1] = *(const u32x4*)(wr[t] + 16);
  }
  for (int k = 0; k < K; k += 64) {
    a1.q[0] = *(const u32x4*)(ar + k + 32);
    a1.q[1] = *(const u32x4*)(ar + k + 48);
    #pragma unroll
    for (int t = 0; t < NT; ++t) {
      b1[t].q[0] = *(const u32x4*)(wr[t] + k + 32);
      b1[t].q[1] = *(const u32x4*)(wr[t] + k + 48);
    }
    #pragma unroll
    for (int t = 0; t < NT; ++t)
      acc[t] = __builtin_amdgcn_wmma_f32_16x16x32_bf16(false, a0.v, false, b0[t].v,
                                                       (short)0, acc[t], false, false);
    a0.q[0] = *(const u32x4*)(ar + k + 64);      // last iter: benign overread
    a0.q[1] = *(const u32x4*)(ar + k + 80);
    #pragma unroll
    for (int t = 0; t < NT; ++t) {
      b0[t].q[0] = *(const u32x4*)(wr[t] + k + 64);
      b0[t].q[1] = *(const u32x4*)(wr[t] + k + 80);
    }
    #pragma unroll
    for (int t = 0; t < NT; ++t)
      acc[t] = __builtin_amdgcn_wmma_f32_16x16x32_bf16(false, a1.v, false, b1[t].v,
                                                       (short)0, acc[t], false, false);
  }
}

// ---------------- conversion / init ----------------------------------------
__global__ void k_convert(const float* __restrict__ s, u16* __restrict__ d, int n) {
  for (int i = blockIdx.x * blockDim.x + threadIdx.x; i < n; i += gridDim.x * blockDim.x)
    d[i] = f2bf(s[i]);
}

__global__ void k_init_state(const float* __restrict__ eh, const float* __restrict__ ec,
                             const float* __restrict__ emb,
                             float* __restrict__ h, float* __restrict__ c,
                             u16* __restrict__ h_bf, u16* __restrict__ x_bf) {
  int i = blockIdx.x * blockDim.x + threadIdx.x;     // B_*H_ threads
  int b = i >> 10, j = i & (H_ - 1);
  float hv = eh[i];
  h[i] = hv;
  c[i] = ec[i];
  h_bf[i] = f2bf(hv);
  x_bf[b * 2 * H_ + j] = f2bf(emb[1 * H_ + j]);      // SOS token id = 1
}

// ---------------- K/V projection: [B*S,H] x [H,H]^T -> bf16 (2 tiles/wave) -
__global__ void k_proj_kv(const u16* __restrict__ A, const u16* __restrict__ W,
                          const float* __restrict__ bias, u16* __restrict__ out) {
  int wid = (blockIdx.x * blockDim.x + threadIdx.x) >> 5;   // 512 * 32 wave-jobs
  int m0 = (wid >> 5) * 16;
  int n0 = (wid & 31) * 32;
  v8f acc[2] = {};
  wmma_mt<2>(A, H_, W, H_, m0, n0, H_, acc);
  int lane = threadIdx.x & 31, hi = lane >> 4, l = lane & 15;
  #pragma unroll
  for (int t = 0; t < 2; ++t) {
    int col = n0 + t * 16 + l;
    float bv = bias[col];
    #pragma unroll
    for (int r = 0; r < 8; ++r) {
      int row = m0 + r + 8 * hi;
      out[(size_t)row * H_ + col] = f2bf(acc[t][r] + bv);
    }
  }
}

// ---------------- q = h @ Wq^T + bq : [16,1024] (2 tiles/wave) -------------
__global__ void k_q_gemm(const u16* __restrict__ h_bf, const u16* __restrict__ Wq,
                         const float* __restrict__ bq, float* __restrict__ q) {
  int wid = (blockIdx.x * blockDim.x + threadIdx.x) >> 5;   // 32 wave-jobs
  int n0 = wid * 32;
  v8f acc[2] = {};
  wmma_mt<2>(h_bf, H_, Wq, H_, 0, n0, H_, acc);
  int lane = threadIdx.x & 31, hi = lane >> 4, l = lane & 15;
  #pragma unroll
  for (int t = 0; t < 2; ++t) {
    int col = n0 + t * 16 + l;
    float bv = bq[col];
    #pragma unroll
    for (int r = 0; r < 8; ++r) q[(r + 8 * hi) * H_ + col] = acc[t][r] + bv;
  }
}

// ---------------- scores + softmax over S, one block per batch -------------
__global__ void k_scores(const float* __restrict__ q, const u16* __restrict__ Kp,
                         float* __restrict__ wgt) {
  __shared__ float sm[512];
  int b = blockIdx.x, s = threadIdx.x;
  const float* qb = q + b * H_;
  const u16* kr = Kp + ((size_t)b * S_ + s) * H_;
  float acc = 0.f;
  for (int j = 0; j < H_; j += 8) {
    u32x4 pk = *(const u32x4*)(kr + j);
    const u16* pp = (const u16*)&pk;
    #pragma unroll
    for (int u = 0; u < 8; ++u) acc += qb[j + u] * bf2f(pp[u]);
  }
  acc *= 0.03125f;                                   // 1/sqrt(1024)
  sm[s] = acc; __syncthreads();
  for (int off = 256; off > 0; off >>= 1) {
    if (s < off) sm[s] = fmaxf(sm[s], sm[s + off]);
    __syncthreads();
  }
  float mx = sm[0]; __syncthreads();
  float e = __expf(acc - mx);
  sm[s] = e; __syncthreads();
  for (int off = 256; off > 0; off >>= 1) {
    if (s < off) sm[s] += sm[s + off];
    __syncthreads();
  }
  wgt[b * S_ + s] = e / sm[0];
}

// ---------------- ctx = w @ V, written into second half of x_bf ------------
__global__ void k_ctx(const float* __restrict__ wgt, const u16* __restrict__ Vp,
                      u16* __restrict__ x_bf) {
  int i = blockIdx.x * blockDim.x + threadIdx.x;     // B_*H_ threads
  int b = i >> 10, hh = i & (H_ - 1);
  const float* wb = wgt + b * S_;
  const u16* vb = Vp + (size_t)b * S_ * H_ + hh;
  float acc = 0.f;
  for (int s = 0; s < S_; ++s) acc += wb[s] * bf2f(vb[(size_t)s * H_]);
  x_bf[b * 2 * H_ + H_ + hh] = f2bf(acc);
}

// ---------------- gates = x@Wih^T + h@Whh^T + biases : [16,4096] -----------
__global__ void k_gates(const u16* __restrict__ x_bf, const u16* __restrict__ Wih,
                        const u16* __restrict__ h_bf, const u16* __restrict__ Whh,
                        const float* __restrict__ bih, const float* __restrict__ bhh,
                        float* __restrict__ gates) {
  int wid = (blockIdx.x * blockDim.x + threadIdx.x) >> 5;   // 128 wave-jobs
  int n0 = wid * 32;
  v8f acc[2] = {};
  wmma_mt<2>(x_bf, 2 * H_, Wih, 2 * H_, 0, n0, 2 * H_, acc);
  wmma_mt<2>(h_bf, H_, Whh, H_, 0, n0, H_, acc);
  int lane = threadIdx.x & 31, hi = lane >> 4, l = lane & 15;
  #pragma unroll
  for (int t = 0; t < 2; ++t) {
    int col = n0 + t * 16 + l;
    float bv = bih[col] + bhh[col];
    #pragma unroll
    for (int r = 0; r < 8; ++r) gates[(r + 8 * hi) * 4 * H_ + col] = acc[t][r] + bv;
  }
}

// ---------------- LSTM cell elementwise ------------------------------------
__global__ void k_lstm(const float* __restrict__ gates, float* __restrict__ h,
                       float* __restrict__ c, u16* __restrict__ h_bf) {
  int idx = blockIdx.x * blockDim.x + threadIdx.x;   // B_*H_ threads
  int b = idx >> 10, j = idx & (H_ - 1);
  const float* gb = gates + b * 4 * H_;
  float gi = gb[j], gf = gb[H_ + j], gg = gb[2 * H_ + j], go = gb[3 * H_ + j];
  float c2 = sigm(gf) * c[idx] + sigm(gi) * tanhf(gg);
  float h2 = sigm(go) * tanhf(c2);
  c[idx] = c2;
  h[idx] = h2;
  h_bf[idx] = f2bf(h2);
}

// ---------------- logits = h2 @ Wout^T + b_out -> d_out[b,t,:] -------------
__global__ void k_logits(const u16* __restrict__ h_bf, const u16* __restrict__ Wout,
                         const float* __restrict__ bout, float* __restrict__ out, int t) {
  int wid = (blockIdx.x * blockDim.x + threadIdx.x) >> 5;   // 1000 wave-jobs
  int n0 = wid * 32;
  v8f acc[2] = {};
  wmma_mt<2>(h_bf, H_, Wout, H_, 0, n0, H_, acc);
  int lane = threadIdx.x & 31, hi = lane >> 4, l = lane & 15;
  #pragma unroll
  for (int tt = 0; tt < 2; ++tt) {
    int col = n0 + tt * 16 + l;
    float bv = bout[col];
    #pragma unroll
    for (int r = 0; r < 8; ++r) {
      int b = r + 8 * hi;
      out[((size_t)b * T_ + t) * V_ + col] = acc[tt][r] + bv;
    }
  }
}

// ---------------- log_softmax over V, one block per (b,t) row --------------
__global__ void k_logsoftmax(float* __restrict__ out) {
  __shared__ float sm[256];
  size_t base = (size_t)blockIdx.x * V_;
  int tid = threadIdx.x;
  float mx = -1e30f;
  for (int i = tid; i < V_; i += 256) mx = fmaxf(mx, out[base + i]);
  sm[tid] = mx; __syncthreads();
  for (int off = 128; off > 0; off >>= 1) {
    if (tid < off) sm[tid] = fmaxf(sm[tid], sm[tid + off]);
    __syncthreads();
  }
  mx = sm[0]; __syncthreads();
  float sum = 0.f;
  for (int i = tid; i < V_; i += 256) sum += __expf(out[base + i] - mx);
  sm[tid] = sum; __syncthreads();
  for (int off = 128; off > 0; off >>= 1) {
    if (tid < off) sm[tid] += sm[tid + off];
    __syncthreads();
  }
  float lse = mx + __logf(sm[0]);
  for (int i = tid; i < V_; i += 256) out[base + i] -= lse;
}

// ---------------- final hT, cT copy ----------------------------------------
__global__ void k_state_out(const float* __restrict__ h, const float* __restrict__ c,
                            float* __restrict__ tail) {
  int i = blockIdx.x * blockDim.x + threadIdx.x;     // B_*H_ threads
  tail[i] = h[i];
  tail[B_ * H_ + i] = c[i];
}

// ===========================================================================
extern "C" void kernel_launch(void* const* d_in, const int* in_sizes, int n_in,
                              void* d_out, int out_size, void* d_ws, size_t ws_size,
                              hipStream_t stream) {
  const float* enc  = (const float*)d_in[0];   // [B,S,H]
  const float* eh   = (const float*)d_in[1];   // [1,B,H]
  const float* ec   = (const float*)d_in[2];   // [1,B,H]
  const float* emb  = (const float*)d_in[3];   // [V,H]
  const float* Wq   = (const float*)d_in[4];
  const float* bq   = (const float*)d_in[5];
  const float* Wk   = (const float*)d_in[6];
  const float* bk   = (const float*)d_in[7];
  const float* Wv   = (const float*)d_in[8];
  const float* bv   = (const float*)d_in[9];
  const float* Wih  = (const float*)d_in[10];  // [4H,2H]
  const float* bih  = (const float*)d_in[11];
  const float* Whh  = (const float*)d_in[12];  // [4H,H]
  const float* bhh  = (const float*)d_in[13];
  const float* Wout = (const float*)d_in[14];  // [V,H]
  const float* bout = (const float*)d_in[15];
  float* out = (float*)d_out;

  char* base = (char*)d_ws;
  size_t off = 0;
  auto carve = [&](size_t bytes) -> char* {
    char* p = base + off;
    off += (bytes + 255) & ~(size_t)255;
    return p;
  };
  u16*   enc_bf  = (u16*)carve((size_t)B_ * S_ * H_ * 2);
  u16*   Wq_bf   = (u16*)carve((size_t)H_ * H_ * 2);
  u16*   Wk_bf   = (u16*)carve((size_t)H_ * H_ * 2);
  u16*   Wv_bf   = (u16*)carve((size_t)H_ * H_ * 2);
  u16*   Wih_bf  = (u16*)carve((size_t)4 * H_ * 2 * H_ * 2);
  u16*   Whh_bf  = (u16*)carve((size_t)4 * H_ * H_ * 2);
  u16*   Wout_bf = (u16*)carve((size_t)V_ * H_ * 2);
  u16*   Kp_bf   = (u16*)carve((size_t)B_ * S_ * H_ * 2);
  u16*   Vp_bf   = (u16*)carve((size_t)B_ * S_ * H_ * 2);
  u16*   x_bf    = (u16*)carve((size_t)B_ * 2 * H_ * 2);
  u16*   h_bf    = (u16*)carve((size_t)B_ * H_ * 2);
  float* hbuf    = (float*)carve((size_t)B_ * H_ * 4);
  float* cbuf    = (float*)carve((size_t)B_ * H_ * 4);
  float* qbuf    = (float*)carve((size_t)B_ * H_ * 4);
  float* wgt     = (float*)carve((size_t)B_ * S_ * 4);
  float* gates   = (float*)carve((size_t)B_ * 4 * H_ * 4);

  // --- one-time: bf16 weight conversion (keeps working set L2-resident) ---
  k_convert<<<2048, 256, 0, stream>>>(enc,  enc_bf,  B_ * S_ * H_);
  k_convert<<<1024, 256, 0, stream>>>(Wq,   Wq_bf,   H_ * H_);
  k_convert<<<1024, 256, 0, stream>>>(Wk,   Wk_bf,   H_ * H_);
  k_convert<<<1024, 256, 0, stream>>>(Wv,   Wv_bf,   H_ * H_);
  k_convert<<<2048, 256, 0, stream>>>(Wih,  Wih_bf,  4 * H_ * 2 * H_);
  k_convert<<<2048, 256, 0, stream>>>(Whh,  Whh_bf,  4 * H_ * H_);
  k_convert<<<4096, 256, 0, stream>>>(Wout, Wout_bf, V_ * H_);

  k_init_state<<<(B_ * H_) / 256, 256, 0, stream>>>(eh, ec, emb, hbuf, cbuf, h_bf, x_bf);

  // --- K/V projections (WMMA, 2 N-tiles/wave): 512*32 = 16384 wave-jobs ---
  k_proj_kv<<<2048, 256, 0, stream>>>(enc_bf, Wk_bf, bk, Kp_bf);
  k_proj_kv<<<2048, 256, 0, stream>>>(enc_bf, Wv_bf, bv, Vp_bf);

  // --- sequential decoder steps ---
  for (int t = 0; t < T_; ++t) {
    k_q_gemm<<<4, 256, 0, stream>>>(h_bf, Wq_bf, bq, qbuf);            // 32 waves
    k_scores<<<B_, 512, 0, stream>>>(qbuf, Kp_bf, wgt);
    k_ctx<<<(B_ * H_) / 256, 256, 0, stream>>>(wgt, Vp_bf, x_bf);
    k_gates<<<16, 256, 0, stream>>>(x_bf, Wih_bf, h_bf, Whh_bf, bih, bhh, gates); // 128 waves
    k_lstm<<<(B_ * H_) / 256, 256, 0, stream>>>(gates, hbuf, cbuf, h_bf);
    k_logits<<<125, 256, 0, stream>>>(h_bf, Wout_bf, bout, out, t);    // 1000 waves
  }

  k_logsoftmax<<<B_ * T_, 256, 0, stream>>>(out);
  k_state_out<<<(B_ * H_) / 256, 256, 0, stream>>>(hbuf, cbuf, out + (size_t)B_ * T_ * V_);
}
